// VanillaRNN_81819126989272
// MI455X (gfx1250) — matrix operand order repaired and checked by reading, not compiled, not run on hardware
//
#include <hip/hip_runtime.h>

// ---------------------------------------------------------------------------
// Vanilla RNN on MI455X (gfx1250): 512 sequential bf16-WMMA GEMM steps.
//  - h stored transposed [B, H] in bf16: B-fragments and D-stores contiguous.
//  - Per-K-chunk A/B tiles staged in LDS by the Tensor Data Mover
//    (tensor_load_to_lds, triple-buffered, s_wait_tensorcnt + barrier).
//  - TDM pad feature gives 80B LDS row stride for bank-friendly ds_load_b128.
//  - hardware v_tanh_f32 epilogue instead of libm tanhf.
// ---------------------------------------------------------------------------

typedef __attribute__((ext_vector_type(16))) __bf16 v16bf;
typedef __attribute__((ext_vector_type(8)))  float  v8f;
typedef __attribute__((ext_vector_type(4)))  unsigned int uint32x4;
typedef __attribute__((ext_vector_type(8)))  int          int32x8;
typedef __attribute__((ext_vector_type(4)))  int          int32x4;

#define RNN_H 2048
#define RNN_B 1024
#define RNN_S 512
#define RNN_C 10

#define KCHUNK     32
#define LDS_STRIDE 80            // 64B of K data + 16B TDM pad per row
#define TILE_BYTES (128 * LDS_STRIDE)   // 10240 B per tile
#define NBUF       3             // triple buffer: 3 * 2 * 10240 = 61440 B LDS

union Frag16 {
    uint4 u[2];   // 32 bytes = 16 bf16 = 8 VGPRs
    v16bf v;
};

__device__ __forceinline__ unsigned short f32_to_bf16(float f) {
    unsigned int u = __float_as_uint(f);
    u += 0x7FFFu + ((u >> 16) & 1u);        // round-to-nearest-even
    return (unsigned short)(u >> 16);
}

__device__ __forceinline__ float bf16_to_f32(unsigned short u) {
    return __uint_as_float(((unsigned int)u) << 16);
}

__device__ __forceinline__ float tanh_hw(float x) {
#if defined(__has_builtin) && __has_builtin(__builtin_amdgcn_tanhf)
    return __builtin_amdgcn_tanhf(x);
#else
    float r;
    asm("v_tanh_f32 %0, %1\n\tv_nop" : "=v"(r) : "v"(x));  // TRANS hazard: 1 slot
    return r;
#endif
}

__device__ __forceinline__ void wait_dscnt0() {
#if defined(__has_builtin) && __has_builtin(__builtin_amdgcn_s_wait_dscnt)
    __builtin_amdgcn_s_wait_dscnt(0);
#else
    asm volatile("s_wait_dscnt 0x0" ::: "memory");
#endif
}

// --- Tensor Data Mover: 2D tile (32 x 128 rows, bf16) global -> LDS ---------
// Tile rows stride tensor_dim0_stride(=2048) elements in memory; TDM inserts
// 16B pad after every 64B stored -> LDS row stride 80B.
__device__ __forceinline__ void tdm_load_tile(const unsigned short* gbase,
                                              unsigned lds_off,
                                              int tensor_rows) {
    unsigned long long ga = (unsigned long long)(const void*)gbase;
    uint32x4 g0;
    g0.x = 1u;                                   // count=1, user mode
    g0.y = lds_off;                              // lds_addr (bytes)
    g0.z = (unsigned)ga;                         // global_addr[31:0]
    g0.w = ((unsigned)(ga >> 32) & 0x01FFFFFFu)  // global_addr[56:32]
         | (2u << 30);                           // type=2 (image)
    int32x8 g1;
    g1[0] = (1 << 16)      // data_size = 1 -> 2 bytes
          | (1 << 20)      // pad_enable
          | (3 << 22)      // pad_interval: 2^(3+1)=16 DWORDs (64B) before pad
          | (3 << 25);     // pad_amount: 3+1 = 4 DWORDs (16B)
    g1[1] = (int)((RNN_H & 0xFFFFu) << 16);          // tensor_dim0[15:0] @63:48
    g1[2] = (int)(((unsigned)tensor_rows & 0xFFFFu) << 16); // dim0 hi16=0 | tensor_dim1 lo16
    g1[3] = (int)(((unsigned)tensor_rows >> 16) | (KCHUNK << 16)); // dim1 hi | tile_dim0=32
    g1[4] = 128;                                      // tile_dim1 = 128, tile_dim2 = 0
    g1[5] = RNN_H;                                    // tensor_dim0_stride lo32 = 2048
    g1[6] = 0;                                        // stride hi16 | dim1_stride lo16
    g1[7] = 0;
    int32x4 z4 = {0, 0, 0, 0};                        // groups 2/3: 2D tile
    int32x8 z8 = {0, 0, 0, 0, 0, 0, 0, 0};            // trailing operand (unused)
    __builtin_amdgcn_tensor_load_to_lds(g0, g1, z4, z4, z8, 0);
}

// ---- one-time f32 -> bf16 conversion of W_hh --------------------------------
__global__ __launch_bounds__(256) void cvt_f32_to_bf16(const float* __restrict__ src,
                                                       unsigned short* __restrict__ dst,
                                                       int n) {
    int i = blockIdx.x * 256 + threadIdx.x;
    if (i < n) dst[i] = f32_to_bf16(src[i]);
}

// ---- zero-init h0 (bf16 zeros), 16B per thread ------------------------------
__global__ __launch_bounds__(256) void zero_u128(uint4* __restrict__ p, int n16) {
    int i = blockIdx.x * 256 + threadIdx.x;
    if (i < n16) {
        uint4 z; z.x = 0u; z.y = 0u; z.z = 0u; z.w = 0u;
        p[i] = z;
    }
}

// ---- fragment loads from LDS + 16 WMMAs for one K-chunk ---------------------
__device__ __forceinline__ void compute_chunk(const char* aTile, const char* bTile,
                                              int wm, int wn, int laneLo, int g,
                                              v8f (&acc)[4][4]) {
    Frag16 a[4], b[4];
    #pragma unroll
    for (int i = 0; i < 4; ++i) {
        const char* p = aTile + (wm * 64 + i * 16 + laneLo) * LDS_STRIDE + 16 * g;
        a[i].u[0] = *(const uint4*)(p);        // K = g*8 .. +7
        a[i].u[1] = *(const uint4*)(p + 32);   // K = 16+g*8 .. +7
    }
    #pragma unroll
    for (int j = 0; j < 4; ++j) {
        const char* p = bTile + (wn * 64 + j * 16 + laneLo) * LDS_STRIDE + 16 * g;
        b[j].u[0] = *(const uint4*)(p);        // K = g*8 .. +7
        b[j].u[1] = *(const uint4*)(p + 32);   // K = 16+g*8 .. +7
    }
    #pragma unroll
    for (int i = 0; i < 4; ++i)
        #pragma unroll
        for (int j = 0; j < 4; ++j)
            acc[i][j] = __builtin_amdgcn_wmma_f32_16x16x32_bf16(
                false, a[i].v, false, b[j].v, (short)0, acc[i][j], false, false);
}

// ---- one recurrence step: hnext = tanh(Whh @ hcur + Whx * x_t + b_h) --------
// 4 waves (2x2), wave tile 64x64, WG tile 128x128, grid 16x8 = 128 WGs.
__global__ __launch_bounds__(128) void rnn_step_wmma(
        const unsigned short* __restrict__ Whh,   // bf16 [H, H] row-major
        const unsigned short* __restrict__ hcur,  // bf16 [B, H]: (k,n) at n*H+k
        unsigned short*       __restrict__ hnext, // bf16 [B, H]
        const float*          __restrict__ x,     // [B, S]
        const float*          __restrict__ whx,   // [H]
        const float*          __restrict__ bh,    // [H]
        int t)
{
    __shared__ char smem[NBUF * 2 * TILE_BYTES];  // 61440 B

    const int lane   = threadIdx.x & 31;
    const int wave   = threadIdx.x >> 5;
    const int laneLo = lane & 15;
    const int g      = lane >> 4;
    const int wm     = wave >> 1;
    const int wn     = wave & 1;

    const int Mblk = blockIdx.x * 128;
    const int Nblk = blockIdx.y * 128;
    const int M0 = Mblk + wm * 64;
    const int N0 = Nblk + wn * 64;

    const unsigned short* Abase = Whh  + (size_t)Mblk * RNN_H;   // A tile rows
    const unsigned short* Bbase = hcur + (size_t)Nblk * RNN_H;   // B tile rows

    unsigned ldsA[NBUF], ldsB[NBUF];
    #pragma unroll
    for (int s = 0; s < NBUF; ++s) {
        ldsA[s] = (unsigned)(unsigned long long)(const void*)&smem[s * 2 * TILE_BYTES];
        ldsB[s] = ldsA[s] + TILE_BYTES;
    }

    v8f acc[4][4];
    #pragma unroll
    for (int i = 0; i < 4; ++i)
        #pragma unroll
        for (int j = 0; j < 4; ++j)
            acc[i][j] = (v8f){0.f, 0.f, 0.f, 0.f, 0.f, 0.f, 0.f, 0.f};

    const bool issuer = (wave == 0);   // one wave drives the TDM

    if (issuer) {                      // prologue: chunks 0 and 1 in flight
        tdm_load_tile(Abase + 0,      ldsA[0], RNN_H);
        tdm_load_tile(Bbase + 0,      ldsB[0], RNN_B);
        tdm_load_tile(Abase + KCHUNK, ldsA[1], RNN_H);
        tdm_load_tile(Bbase + KCHUNK, ldsB[1], RNN_B);
    }

    const int NCHUNK = RNN_H / KCHUNK;   // 64
    #pragma unroll 1
    for (int c = 0; c < NCHUNK - 2; ++c) {
        wait_dscnt0();                                   // done reading slot (c+2)%3
        if (issuer) __builtin_amdgcn_s_wait_tensorcnt(2); // chunk c landed (c+1 in flight)
        __syncthreads();
        if (issuer) {
            const int kn = (c + 2) * KCHUNK;
            tdm_load_tile(Abase + kn, ldsA[(c + 2) % NBUF], RNN_H);
            tdm_load_tile(Bbase + kn, ldsB[(c + 2) % NBUF], RNN_B);
        }
        const char* base = &smem[(c % NBUF) * 2 * TILE_BYTES];
        compute_chunk(base, base + TILE_BYTES, wm, wn, laneLo, g, acc);
    }
    {   // chunk 62: chunk 63 still in flight
        wait_dscnt0();
        if (issuer) __builtin_amdgcn_s_wait_tensorcnt(2);
        __syncthreads();
        const char* base = &smem[((NCHUNK - 2) % NBUF) * 2 * TILE_BYTES];
        compute_chunk(base, base + TILE_BYTES, wm, wn, laneLo, g, acc);
    }
    {   // chunk 63: drain
        wait_dscnt0();
        if (issuer) __builtin_amdgcn_s_wait_tensorcnt(0);
        __syncthreads();
        const char* base = &smem[((NCHUNK - 1) % NBUF) * 2 * TILE_BYTES];
        compute_chunk(base, base + TILE_BYTES, wm, wn, laneLo, g, acc);
    }

    // Epilogue: pre = acc + whx[m]*x[n,t] + bh[m]; h = tanh(pre); store bf16.
    float wv[4][8], bv[4][8];
    #pragma unroll
    for (int i = 0; i < 4; ++i) {
        const int mb = M0 + i * 16 + g * 8;
        #pragma unroll
        for (int r = 0; r < 8; ++r) {
            wv[i][r] = whx[mb + r];
            bv[i][r] = bh[mb + r];
        }
    }
    #pragma unroll
    for (int j = 0; j < 4; ++j) {
        const int   n   = N0 + j * 16 + laneLo;
        const float xnt = x[(size_t)n * RNN_S + t];
        #pragma unroll
        for (int i = 0; i < 4; ++i) {
            const int mb = M0 + i * 16 + g * 8;
            unsigned int packed[4];
            #pragma unroll
            for (int r = 0; r < 8; ++r) {
                float pre = acc[i][j][r] + wv[i][r] * xnt + bv[i][r];
                unsigned short us = f32_to_bf16(tanh_hw(pre));
                if (r & 1) packed[r >> 1] |= ((unsigned int)us) << 16;
                else       packed[r >> 1]  = (unsigned int)us;
            }
            uint4 st; st.x = packed[0]; st.y = packed[1]; st.z = packed[2]; st.w = packed[3];
            *(uint4*)(hnext + (size_t)n * RNN_H + mb) = st;   // 8 consecutive K
        }
    }
}

// ---- final projection: out[b,c] = b_p[c] + sum_h W_hy[h,c] * h_final[h,b] ---
__global__ __launch_bounds__(256) void rnn_out_proj(
        const unsigned short* __restrict__ hfin,  // bf16 [B, H]
        const float*          __restrict__ why,   // [H, C]
        const float*          __restrict__ bp,    // [C]
        float*                __restrict__ out)   // [B, C]
{
    int idx = blockIdx.x * 256 + threadIdx.x;
    if (idx >= RNN_B * RNN_C) return;
    int b = idx / RNN_C, c = idx % RNN_C;
    const unsigned short* hrow = hfin + (size_t)b * RNN_H;
    float acc = bp[c];
    for (int k = 0; k < RNN_H; ++k)
        acc += bf16_to_f32(hrow[k]) * why[k * RNN_C + c];
    out[idx] = acc;
}

extern "C" void kernel_launch(void* const* d_in, const int* in_sizes, int n_in,
                              void* d_out, int out_size, void* d_ws, size_t ws_size,
                              hipStream_t stream) {
    const float* x   = (const float*)d_in[0];   // [B, S]
    const float* whx = (const float*)d_in[1];   // [H, 1]
    const float* whh = (const float*)d_in[2];   // [H, H]
    const float* why = (const float*)d_in[3];   // [H, C]
    const float* bh  = (const float*)d_in[4];   // [H, 1]
    const float* bp  = (const float*)d_in[5];   // [C, 1]
    float* out = (float*)d_out;

    // Workspace: W_hh bf16 (8 MB) | h ping (4 MB) | h pong (4 MB)
    char* ws = (char*)d_ws;
    unsigned short* whh_bf = (unsigned short*)ws;
    unsigned short* h0     = (unsigned short*)(ws + (size_t)8 * 1024 * 1024);
    unsigned short* h1     = (unsigned short*)(ws + (size_t)12 * 1024 * 1024);

    cvt_f32_to_bf16<<<(RNN_H * RNN_H + 255) / 256, 256, 0, stream>>>(whh, whh_bf, RNN_H * RNN_H);
    const int n16 = (RNN_B * RNN_H * 2) / 16;
    zero_u128<<<(n16 + 255) / 256, 256, 0, stream>>>((uint4*)h0, n16);

    dim3 grid(RNN_H / 128, RNN_B / 128);  // 16 x 8
    dim3 blk(128);
    for (int t = 0; t < RNN_S; ++t) {
        const unsigned short* hc = (t & 1) ? h1 : h0;
        unsigned short*       hn = (t & 1) ? h0 : h1;
        rnn_step_wmma<<<grid, blk, 0, stream>>>(whh_bf, hc, hn, x, whx, bh, t);
    }
    rnn_out_proj<<<(RNN_B * RNN_C + 255) / 256, 256, 0, stream>>>(h0, why, bp, out);
}
